// BiDynamicRNN_39719857553997
// MI455X (gfx1250) — matrix operand order repaired and checked by reading, not compiled, not run on hardware
//
#include <hip/hip_runtime.h>
#include <stdint.h>

typedef unsigned short u16;

#define B_ 32
#define T_ 2048
#define F_ 256
#define H_ 256
#define G4_ 1024      // 4*H
#define KTOT_ 512     // F+H
#define NWG_DIR 8     // workgroups per direction
#define NKT 16        // K tiles of 32

typedef __attribute__((ext_vector_type(16))) __bf16 v16bf;
typedef __attribute__((ext_vector_type(8)))  float  v8f;

struct Frag32B { uint4 lo, hi; };

// ---------------- workspace layout (bytes) ----------------
// [0,256)                 : sync counters (dir0 at +0, dir1 at +128)
// [256, 256+32MB)         : xbf  bf16 [T][B][F]   (t-major)
// [.., +2MB)              : wt   bf16 swizzled B-fragments
// [.., +64KB)             : hbuf bf16 [dir][parity][B][H]
#define XBF_OFF  ((size_t)256)
#define WT_OFF   (XBF_OFF + (size_t)T_ * B_ * F_ * 2)
#define HB_OFF   (WT_OFF + (size_t)2 * NWG_DIR * 8 * NKT * 32 * 16 * 2)

__device__ __forceinline__ u16 f2bf(float f) {
  unsigned u = __float_as_uint(f);
  unsigned r = u + 0x7FFFu + ((u >> 16) & 1u);   // round to nearest even
  return (u16)(r >> 16);
}

__device__ __forceinline__ float sigm(float x) {
  return 1.0f / (1.0f + __expf(-x));
}

// low 32 bits of a generic LDS pointer are the LDS byte offset
__device__ __forceinline__ unsigned lds_off32(const void* p) {
  return (unsigned)(uintptr_t)p;
}

// ---------------- init: zero counters + h buffers ----------------
__global__ __launch_bounds__(256) void init_k(u16* hbuf, int* counters) {
  int i = blockIdx.x * 256 + threadIdx.x;
  // hbuf = 2*2*B*H u16 = 32768 u16 = 16384 uints
  if (i < 16384) ((unsigned*)hbuf)[i] = 0u;
  if (i == 0) { counters[0] = 0; counters[32] = 0; }
}

// ---------------- x fp32 [B][T][F] -> bf16 [T][B][F] ----------------
__global__ __launch_bounds__(256) void conv_x(const float* __restrict__ x,
                                              u16* __restrict__ xbf) {
  int c   = blockIdx.x * 256 + threadIdx.x;  // chunk of 4 floats
  int f4  = c & 63;                          // F/4 = 64
  int rem = c >> 6;
  int t   = rem & (T_ - 1);
  int b   = rem >> 11;
  const float4 v = *(const float4*)(x + ((size_t)b * T_ + t) * F_ + f4 * 4);
  uint2 o;
  o.x = (unsigned)f2bf(v.x) | ((unsigned)f2bf(v.y) << 16);
  o.y = (unsigned)f2bf(v.z) | ((unsigned)f2bf(v.w) << 16);
  *(uint2*)(xbf + ((size_t)t * B_ + b) * F_ + f4 * 4) = o;
}

// ---------------- W fp32 (512,1024) -> swizzled bf16 B-fragments ----------------
// layout: wt[((((d*8+wg)*8+nt)*16+kt)*32+lane)*16+e]
// lane -> column nt*16+(lane&15); K = kt*32 + (lane>>4)*16 + e   (contiguous 16/lane)
__global__ __launch_bounds__(256) void prep_w(const float* __restrict__ Wfw,
                                              const float* __restrict__ Wbw,
                                              u16* __restrict__ wt) {
  unsigned idx = blockIdx.x * 256 + threadIdx.x;   // < 2^20
  int e    = idx & 15;
  int lane = (idx >> 4) & 31;
  int kt   = (idx >> 9) & 15;
  int nt   = (idx >> 13) & 7;
  int wg   = (idx >> 16) & 7;
  int d    = (idx >> 19) & 1;
  int col_local = nt * 16 + (lane & 15);   // 0..127 within WG slice
  int blk  = col_local >> 5;               // gate block i/j/f/o
  int unit = col_local & 31;
  int g    = blk * H_ + wg * 32 + unit;    // global gate column
  int k    = kt * 32 + ((lane >> 4) << 4) + e;
  const float* W = d ? Wbw : Wfw;
  wt[idx] = f2bf(W[(size_t)k * G4_ + g]);
}

// ---------------- persistent bi-LSTM scan ----------------
__global__ __launch_bounds__(256) void lstm_scan(const u16* __restrict__ xbf,
                                                 const u16* __restrict__ wt,
                                                 const float* __restrict__ bias_fw,
                                                 const float* __restrict__ bias_bw,
                                                 u16* hbuf, int* counters,
                                                 float* __restrict__ out) {
  __shared__ u16   Abuf[32][KTOT_];   // 32 KB : [x_t | h_{t-1}]
  __shared__ float gbuf[32][128];     // 16 KB : gate pre-activations
  __shared__ float cstate[32][32];    //  4 KB : cell state slice

  const int tid  = threadIdx.x;
  const int wave = tid >> 5;
  const int lane = tid & 31;
  const int d    = blockIdx.x >> 3;   // direction
  const int wg   = blockIdx.x & 7;    // hidden-unit slice

  const float* bias = d ? bias_bw : bias_fw;
  int* cnt = counters + d * 32;                    // 128B apart
  u16* hb  = hbuf + (size_t)d * (2 * B_ * H_);

  for (int i = tid; i < 32 * 32; i += 256) ((float*)cstate)[i] = 0.0f;

  // this wave's N-tile; preload all 16 B-fragments into registers (stay
  // resident in VGPRs for the entire 2048-step scan)
  const int nt = wave;
  const u16* wtbase =
      wt + (((size_t)(d * NWG_DIR + wg) * 8 + nt) * NKT) * 32 * 16;
  v16bf bfrag[NKT];
#pragma unroll
  for (int kt = 0; kt < NKT; ++kt) {
    Frag32B fb;
    const uint4* bp = (const uint4*)(wtbase + ((size_t)kt * 32 + lane) * 16);
    fb.lo = bp[0];
    fb.hi = bp[1];
    bfrag[kt] = __builtin_bit_cast(v16bf, fb);
  }

  // per-thread fixed hidden unit for the gate phase
  const int u0 = tid & 31;
  const float bi = bias[0 * H_ + wg * 32 + u0];
  const float bj = bias[1 * H_ + wg * 32 + u0];
  const float bf = bias[2 * H_ + wg * 32 + u0];
  const float bo = bias[3 * H_ + wg * 32 + u0];

  const int arow = lane & 15;
  const int ksub = (lane >> 4) << 4;

  // per-thread staging geometry (8-u16 = 16B chunks, 32 chunks per 256-col row)
  const int srow = tid >> 5;            // rows 0..7 base, +8 per iteration
  const int scol = (tid & 31) << 3;     // cols 0..248

  for (int t = 0; t < T_; ++t) {
    // ---- 1. wait for all WGs of this direction to finish step t-1 ----
    if (t > 0 && tid == 0) {
      const int target = NWG_DIR * t;
      while (__hip_atomic_load(cnt, __ATOMIC_ACQUIRE,
                               __HIP_MEMORY_SCOPE_AGENT) < target)
        __builtin_amdgcn_s_sleep(1);
    }
    __syncthreads();

    // ---- 2. async-stage A = [x_t | h_{t-1}] into LDS (memory -> LDS, no VGPRs) ----
    const int tx = d ? (T_ - 1 - t) : t;
    const u16* xrow = xbf + (size_t)tx * (B_ * F_);
    const u16* hrow = hb + ((t + 1) & 1) * (B_ * H_);
#pragma unroll
    for (int it = 0; it < 4; ++it) {
      int row = srow + it * 8;
      unsigned lds  = lds_off32(&Abuf[row][scol]);
      unsigned goff = (unsigned)((row * F_ + scol) * 2);
      asm volatile("global_load_async_to_lds_b128 %0, %1, %2"
                   :
                   : "v"(lds), "v"(goff), "s"(xrow)
                   : "memory");
    }
#pragma unroll
    for (int it = 0; it < 4; ++it) {
      int row = srow + it * 8;
      unsigned lds  = lds_off32(&Abuf[row][F_ + scol]);
      unsigned goff = (unsigned)((row * H_ + scol) * 2);
      asm volatile("global_load_async_to_lds_b128 %0, %1, %2"
                   :
                   : "v"(lds), "v"(goff), "s"(hrow)
                   : "memory");
    }
    // prefetch next time-step's x tile while this one lands
    if (t + 1 < T_) {
      const u16* nx = xbf + (size_t)(d ? (T_ - 2 - t) : (t + 1)) * (B_ * F_);
      __builtin_prefetch(nx + tid * 32, 0, 1);
    }
    asm volatile("s_wait_asynccnt 0" ::: "memory");
    __syncthreads();

    // ---- 3. WMMA: gates[32 x 128]; this wave owns cols [nt*16, nt*16+16) ----
    v8f acc0 = {}, acc1 = {};
#pragma unroll
    for (int kt = 0; kt < NKT; ++kt) {
      const int kb = kt * 32 + ksub;
      Frag32B fa0, fa1;
      fa0.lo = *(const uint4*)(&Abuf[arow][kb]);
      fa0.hi = *(const uint4*)(&Abuf[arow][kb + 8]);
      fa1.lo = *(const uint4*)(&Abuf[16 + arow][kb]);
      fa1.hi = *(const uint4*)(&Abuf[16 + arow][kb + 8]);
      v16bf a0 = __builtin_bit_cast(v16bf, fa0);
      v16bf a1 = __builtin_bit_cast(v16bf, fa1);
      acc0 = __builtin_amdgcn_wmma_f32_16x16x32_bf16(false, a0, false, bfrag[kt],
                                                     (short)0, acc0, false, false);
      acc1 = __builtin_amdgcn_wmma_f32_16x16x32_bf16(false, a1, false, bfrag[kt],
                                                     (short)0, acc1, false, false);
    }

    // ---- 4. accumulators -> LDS gate buffer ----
    {
      const int colb = nt * 16 + (lane & 15);
      const int rb   = (lane >> 4) << 3;
#pragma unroll
      for (int r = 0; r < 8; ++r) {
        gbuf[rb + r][colb]      = acc0[r];
        gbuf[16 + rb + r][colb] = acc1[r];
      }
    }
    __syncthreads();

    // ---- 5. gate nonlinearities + state update ----
    u16* hdst = hb + (t & 1) * (B_ * H_);
    const int tout = d ? (T_ - 1 - t) : t;
#pragma unroll
    for (int i = 0; i < 4; ++i) {
      int b = (tid >> 5) + i * 8;
      float gi = gbuf[b][u0]      + bi;
      float gj = gbuf[b][32 + u0] + bj;
      float gf = gbuf[b][64 + u0] + bf;
      float go = gbuf[b][96 + u0] + bo;
      float c  = cstate[b][u0];
      float cn = c * sigm(gf + 1.0f) + sigm(gi) * tanhf(gj);
      float h  = tanhf(cn) * sigm(go);
      cstate[b][u0] = cn;
      hdst[b * H_ + wg * 32 + u0] = f2bf(h);
      out[((size_t)b * T_ + tout) * (2 * H_) + d * H_ + wg * 32 + u0] = h;
    }

    // ---- 6. signal step complete (release) ----
    __threadfence();
    __syncthreads();
    if (tid == 0)
      __hip_atomic_fetch_add(cnt, 1, __ATOMIC_RELEASE, __HIP_MEMORY_SCOPE_AGENT);
  }
}

// ---------------- host launcher ----------------
extern "C" void kernel_launch(void* const* d_in, const int* in_sizes, int n_in,
                              void* d_out, int out_size, void* d_ws, size_t ws_size,
                              hipStream_t stream) {
  const float* x    = (const float*)d_in[0];
  const float* W_fw = (const float*)d_in[1];
  const float* b_fw = (const float*)d_in[2];
  const float* W_bw = (const float*)d_in[3];
  const float* b_bw = (const float*)d_in[4];
  float* out = (float*)d_out;

  char* ws      = (char*)d_ws;
  int*  cnts    = (int*)ws;
  u16*  xbf     = (u16*)(ws + XBF_OFF);
  u16*  wt      = (u16*)(ws + WT_OFF);
  u16*  hbuf    = (u16*)(ws + HB_OFF);

  hipLaunchKernelGGL(init_k, dim3(64),    dim3(256), 0, stream, hbuf, cnts);
  hipLaunchKernelGGL(conv_x, dim3(16384), dim3(256), 0, stream, x, xbf);
  hipLaunchKernelGGL(prep_w, dim3(4096),  dim3(256), 0, stream, W_fw, W_bw, wt);
  hipLaunchKernelGGL(lstm_scan, dim3(16), dim3(256), 0, stream,
                     xbf, wt, b_fw, b_bw, hbuf, cnts, out);
}
